// MlpMoeBlock_13048110645666
// MI455X (gfx1250) — matrix operand order repaired and checked by reading, not compiled, not run on hardware
//
#include <hip/hip_runtime.h>
#include <hip/hip_bf16.h>
#include <stdint.h>

// CDNA5 (gfx1250) wave32 WMMA types
typedef __attribute__((ext_vector_type(16))) __bf16    v16bf;
typedef __attribute__((ext_vector_type(2)))  __bf16    v2bf;
typedef __attribute__((ext_vector_type(8)))  float     v8f;
typedef __attribute__((ext_vector_type(4)))  uint32_t  u32x4;
typedef __attribute__((ext_vector_type(2)))  uint32_t  u32x2;

#define LDA 40   // padded LDS row stride in bf16 elems (80B: 16B aligned, conflict-free)

// pack two floats as bf16 pair (RNE) into one dword: lo = a, hi = b
#if __has_builtin(__builtin_amdgcn_cvt_pk_bf16_f32)
__device__ __forceinline__ uint32_t pack_bf16x2(float a, float b) {
    union { v2bf v; uint32_t u; } c;
    c.v = __builtin_amdgcn_cvt_pk_bf16_f32(a, b);
    return c.u;
}
#else
__device__ __forceinline__ uint32_t pack_bf16x2(float a, float b) {
    union { float f; uint32_t u; } xa, xb; xa.f = a; xb.f = b;
    const uint32_t ra = xa.u + 0x7FFFu + ((xa.u >> 16) & 1u);
    const uint32_t rb = xb.u + 0x7FFFu + ((xb.u >> 16) & 1u);
    return (ra >> 16) | (rb & 0xFFFF0000u);
}
#endif

__device__ __forceinline__ uint16_t f32_bf16_rne(float f) {
    return (uint16_t)pack_bf16x2(f, 0.0f);
}

union FragAB { u32x4 q[2]; v16bf v; };

// ---- gfx1250 async global->LDS copy (ASYNCcnt-tracked, no VGPR data) ----
#if __has_builtin(__builtin_amdgcn_global_load_async_to_lds_b64)
typedef int v2i_vs __attribute__((vector_size(8)));
typedef __attribute__((address_space(1))) v2i_vs* as1_v2i_p;
typedef __attribute__((address_space(3))) v2i_vs* as3_v2i_p;
__device__ __forceinline__ void async_cp_b64(const void* g, void* l) {
    __builtin_amdgcn_global_load_async_to_lds_b64(
        (as1_v2i_p)(uintptr_t)g,
        (as3_v2i_p)(uint32_t)(uintptr_t)l, 0, 0);
}
#else
__device__ __forceinline__ void async_cp_b64(const void* g, void* l) {
    *(u32x2*)l = *(const u32x2*)g;   // fallback: synchronous copy through VGPRs
}
#endif

__device__ __forceinline__ void wait_async0() {
#if __has_builtin(__builtin_amdgcn_s_wait_asynccnt)
    __builtin_amdgcn_s_wait_asynccnt(0);
#else
    asm volatile("s_wait_asynccnt 0x0" ::: "memory");
#endif
}

// ---------------------------------------------------------------------------
// Router: one wave per token. gates[t][e] = softmax_e( x[t,:] . wr[:,e] )
// ---------------------------------------------------------------------------
__global__ __launch_bounds__(256) void router_kernel(const float* __restrict__ x,
                                                     const float* __restrict__ wr,
                                                     float* __restrict__ gates,
                                                     int H) {
    const int lane = threadIdx.x & 31;
    const int wave = threadIdx.x >> 5;
    const int t = blockIdx.x * 8 + wave;
    const float* xp = x + (size_t)t * H;

    float acc[8];
#pragma unroll
    for (int e = 0; e < 8; ++e) acc[e] = 0.0f;

    for (int h = lane; h < H; h += 32) {
        const float xv = xp[h];
        const float* w = wr + h * 8;
#pragma unroll
        for (int e = 0; e < 8; ++e) acc[e] += xv * w[e];
    }
#pragma unroll
    for (int e = 0; e < 8; ++e) {
#pragma unroll
        for (int off = 16; off > 0; off >>= 1)
            acc[e] += __shfl_xor(acc[e], off, 32);
    }
    float mx = acc[0];
#pragma unroll
    for (int e = 1; e < 8; ++e) mx = fmaxf(mx, acc[e]);
    float s = 0.0f;
#pragma unroll
    for (int e = 0; e < 8; ++e) { acc[e] = __expf(acc[e] - mx); s += acc[e]; }
    const float inv = 1.0f / s;
    if (lane < 8) gates[(size_t)t * 8 + lane] = acc[lane] * inv;
}

// ---------------------------------------------------------------------------
// Elementwise f32 -> bf16 (RNE), 4 elems/thread
// ---------------------------------------------------------------------------
__global__ __launch_bounds__(256) void cvt_bf16_kernel(const float* __restrict__ src,
                                                       uint16_t* __restrict__ dst) {
    const size_t i4 = ((size_t)blockIdx.x * 256 + threadIdx.x) * 4;
    const float4 f = *(const float4*)(src + i4);
    u32x2 t;
    t.x = pack_bf16x2(f.x, f.y);
    t.y = pack_bf16x2(f.z, f.w);
    *(u32x2*)(dst + i4) = t;
}

// ---------------------------------------------------------------------------
// Tiled transpose + convert: src [K][N] f32 -> dst [N][K] bf16 (per-expert z)
// ---------------------------------------------------------------------------
__global__ __launch_bounds__(256) void transpose_bf16_kernel(const float* __restrict__ src,
                                                             uint16_t* __restrict__ dst,
                                                             int K, int N) {
    __shared__ __align__(16) uint16_t T[32 * 36];   // stride 36 elems (72B rows)
    const int tid = threadIdx.x;
    const size_t eoff = (size_t)blockIdx.z * K * N;
    const int n0 = blockIdx.x * 32;
    const int k0 = blockIdx.y * 32;

#pragma unroll
    for (int i = 0; i < 4; ++i) {
        const int idx = tid + (i << 8);
        const int kr = idx >> 5;      // 0..31
        const int nc = idx & 31;      // 0..31
        const float v = src[eoff + (size_t)(k0 + kr) * N + n0 + nc];
        T[nc * 36 + kr] = f32_bf16_rne(v);
    }
    __syncthreads();
    const int n  = tid >> 3;          // 0..31
    const int kc = (tid & 7) << 2;    // 0..28
    const u32x2 t = *(const u32x2*)&T[n * 36 + kc];
    *(u32x2*)(dst + eoff + (size_t)(n0 + n) * K + k0 + kc) = t;
}

// ---------------------------------------------------------------------------
// Tiled bf16 WMMA GEMM, 256x128 block tile, K-tile 32, 256 threads (8 waves,
// 4x2 grid of 64x64 wave tiles => 4x4 WMMA accumulators per wave).
// Double-buffered LDS + software pipelining (1 barrier per K-step).
// PRE=1: A bf16 [M][K], B bf16 pre-transposed [N][K]; ALL staging via
//        global_load_async_to_lds_b64 (no staging VALU at all).
// PRE=0: fallback, B f32 [K][N] converted in-kernel; MODE0 A f32 via regs.
// MODE 0: fc1  C = gelu(A*B + b1) -> hid (bf16)
// MODE 1: fc2  out (+)= gate * (A*B + b2)
// ---------------------------------------------------------------------------
template <int MODE, bool PRE>
__global__ __launch_bounds__(256) void moe_gemm(const void*  __restrict__ Aglob,
                                                const void*  __restrict__ Bglob,
                                                const float* __restrict__ bias,
                                                const float* __restrict__ gates,
                                                uint16_t*    __restrict__ hidOut,
                                                float*       __restrict__ out,
                                                int K, int N, int expert) {
    __shared__ __align__(16) uint16_t Asl[2][256 * LDA];
    __shared__ __align__(16) uint16_t Bsl[2][128 * LDA];

    const int tid  = threadIdx.x;
    const int lane = tid & 31;
    const int wave = tid >> 5;
    const int wr_  = wave & 3;     // 4 wave-rows, 64 M each
    const int wc_  = wave >> 2;    // 2 wave-cols, 64 N each
    const int rowBlk = blockIdx.x * 256;
    const int colBlk = blockIdx.y * 128;

    const int half = lane >> 4;    // wave32 lane half (ISA fragment layout)
    const int l15  = lane & 15;

    // staging assignments (per thread): rows r0 + 32*i, 8B k-chunk at cc
    const int r0 = tid >> 3;           // 0..31
    const int cc = (tid & 7) << 2;     // 0..28 (bf16 elems)
    const int bN  = (tid & 31) << 2;   // PRE=0 B: n quad (coalesced along n)
    const int bKp = tid >> 5;          // PRE=0 B: k-pair: k = 2*(bKp + 8*i)

    v8f acc[4][4];
#pragma unroll
    for (int r = 0; r < 4; ++r)
#pragma unroll
        for (int c = 0; c < 4; ++c)
#pragma unroll
            for (int i = 0; i < 8; ++i) acc[r][c][i] = 0.0f;

    float4 aReg[8];
    float4 bReg[4];
    const int nKT = K >> 5;

    // ---------------- PRE=1 staging: pure async bf16 copies ----------------
    auto stageA_async = [&](int kt, int buf) {   // 256 rows
        const uint16_t* Ab = (const uint16_t*)Aglob;
#pragma unroll
        for (int i = 0; i < 8; ++i)
            async_cp_b64(Ab + (size_t)(rowBlk + r0 + 32 * i) * K + (kt << 5) + cc,
                         &Asl[buf][(r0 + 32 * i) * LDA + cc]);
    };
    auto stageB_async = [&](int kt, int buf) {   // B is [N][K] bf16, 128 rows
        const uint16_t* Bt = (const uint16_t*)Bglob;
#pragma unroll
        for (int i = 0; i < 4; ++i)
            async_cp_b64(Bt + (size_t)(colBlk + r0 + 32 * i) * K + (kt << 5) + cc,
                         &Bsl[buf][(r0 + 32 * i) * LDA + cc]);
    };
    // ---------------- PRE=0 staging: convert in-kernel ---------------------
    auto loadB = [&](int kt) {
        const float* B = (const float*)Bglob;
#pragma unroll
        for (int i = 0; i < 2; ++i) {
            const int k = ((bKp + 8 * i) << 1);
            bReg[2 * i + 0] = *(const float4*)(B + (size_t)((kt << 5) + k + 0) * N + colBlk + bN);
            bReg[2 * i + 1] = *(const float4*)(B + (size_t)((kt << 5) + k + 1) * N + colBlk + bN);
        }
    };
    auto storeB = [&](int buf) {
#pragma unroll
        for (int i = 0; i < 2; ++i) {
            const int k = ((bKp + 8 * i) << 1);
            const float4 f0 = bReg[2 * i + 0];
            const float4 f1 = bReg[2 * i + 1];
            *(uint32_t*)&Bsl[buf][(bN + 0) * LDA + k] = pack_bf16x2(f0.x, f1.x);
            *(uint32_t*)&Bsl[buf][(bN + 1) * LDA + k] = pack_bf16x2(f0.y, f1.y);
            *(uint32_t*)&Bsl[buf][(bN + 2) * LDA + k] = pack_bf16x2(f0.z, f1.z);
            *(uint32_t*)&Bsl[buf][(bN + 3) * LDA + k] = pack_bf16x2(f0.w, f1.w);
        }
    };
    auto loadA = [&](int kt) {
        const float* A = (const float*)Aglob;
#pragma unroll
        for (int i = 0; i < 8; ++i)
            aReg[i] = *(const float4*)(A + (size_t)(rowBlk + r0 + 32 * i) * K + (kt << 5) + cc);
    };
    auto storeA = [&](int buf) {
#pragma unroll
        for (int i = 0; i < 8; ++i) {
            u32x2 t;
            t.x = pack_bf16x2(aReg[i].x, aReg[i].y);
            t.y = pack_bf16x2(aReg[i].z, aReg[i].w);
            *(u32x2*)&Asl[buf][(r0 + 32 * i) * LDA + cc] = t;
        }
    };
    auto compute = [&](int buf) {
        FragAB a[4], b[4];
#pragma unroll
        for (int r = 0; r < 4; ++r) {
            const uint16_t* p = &Asl[buf][(wr_ * 64 + r * 16 + l15) * LDA + half * 8];
            a[r].q[0] = *(const u32x4*)p;          // K = half*8 + 0..7
            a[r].q[1] = *(const u32x4*)(p + 16);   // K = 16 + half*8 + 0..7
        }
#pragma unroll
        for (int c = 0; c < 4; ++c) {
            const uint16_t* p = &Bsl[buf][(wc_ * 64 + c * 16 + l15) * LDA + half * 8];
            b[c].q[0] = *(const u32x4*)p;          // K = half*8 + 0..7
            b[c].q[1] = *(const u32x4*)(p + 16);   // K = 16 + half*8 + 0..7
        }
#pragma unroll
        for (int r = 0; r < 4; ++r)
#pragma unroll
            for (int c = 0; c < 4; ++c)
                acc[r][c] = __builtin_amdgcn_wmma_f32_16x16x32_bf16(
                    false, a[r].v, false, b[c].v, (short)0, acc[r][c], false, false);
    };

    // ---- prologue: stage tile 0 into buffer 0 ----
    if (PRE) {
        stageA_async(0, 0); stageB_async(0, 0);
    } else if (MODE == 0) {
        loadA(0); loadB(0); storeA(0); storeB(0);
    } else {
        stageA_async(0, 0); loadB(0); storeB(0);
    }

    int buf = 0;
    for (int kt = 0; kt < nKT; ++kt) {
        if (PRE || MODE == 1) wait_async0();   // this wave's async writes done
        __syncthreads();                       // all waves' staging of `buf` done
        const int  nb   = buf ^ 1;
        const bool more = (kt + 1) < nKT;
        if (more) {                            // issue next-tile staging early
            if (PRE)            { stageA_async(kt + 1, nb); stageB_async(kt + 1, nb); }
            else if (MODE == 0) { loadA(kt + 1); loadB(kt + 1); }
            else                { stageA_async(kt + 1, nb); loadB(kt + 1); }
        }
        compute(buf);                          // 16x ds_load_b128 + 16x v_wmma
        if (!PRE && more) {
            if (MODE == 0) storeA(nb);
            storeB(nb);
        }
        buf = nb;
    }

    // ---- epilogue ----
#pragma unroll
    for (int r = 0; r < 4; ++r) {
#pragma unroll
        for (int c = 0; c < 4; ++c) {
            const int n  = colBlk + wc_ * 64 + c * 16 + l15;
            const float bv = bias[n];
#pragma unroll
            for (int i = 0; i < 8; ++i) {
                const int m = rowBlk + wr_ * 64 + r * 16 + half * 8 + i;
                float v = acc[r][c][i] + bv;
                if (MODE == 0) {
                    v = 0.5f * v * (1.0f + erff(v * 0.70710678118654752f));
                    hidOut[(size_t)m * N + n] = f32_bf16_rne(v);
                } else {
                    const float g = gates[(size_t)m * 8 + expert];
                    const float o = g * v;
                    float* dst = &out[(size_t)m * N + n];
                    if (expert == 0) *dst = o;
                    else             *dst += o;
                }
            }
        }
    }
}

// ---------------------------------------------------------------------------
extern "C" void kernel_launch(void* const* d_in, const int* in_sizes, int n_in,
                              void* d_out, int out_size, void* d_ws, size_t ws_size,
                              hipStream_t stream) {
    (void)in_sizes; (void)n_in; (void)out_size;
    const float* x  = (const float*)d_in[0];   // [8192,1024]
    const float* w1 = (const float*)d_in[1];   // [8,1024,4096]
    const float* b1 = (const float*)d_in[2];   // [8,4096]
    const float* w2 = (const float*)d_in[3];   // [8,4096,1024]
    const float* b2 = (const float*)d_in[4];   // [8,1024]
    const float* wr = (const float*)d_in[5];   // [1024,8]
    float* out = (float*)d_out;                // [8192,1024]

    const int Tok = 8192, H = 1024, M = 4096, E = 8;

    // workspace layout (all offsets 256B-aligned)
    const size_t offGates = 0;                                   // 256 KB
    const size_t offHid   = (size_t)Tok * E * sizeof(float);     // + 64 MB (bf16 hid)
    const size_t offXb    = offHid + (size_t)Tok * M * 2;        // + 16 MB (bf16 x)
    const size_t offW1t   = offXb  + (size_t)Tok * H * 2;        // + 64 MB (bf16 w1^T)
    const size_t offW2t   = offW1t + (size_t)E * H * M * 2;      // + 64 MB (bf16 w2^T)
    const size_t needFull = offW2t + (size_t)E * M * H * 2;      // ~208 MB total

    float*    gates = (float*)((char*)d_ws + offGates);
    uint16_t* hid   = (uint16_t*)((char*)d_ws + offHid);

    router_kernel<<<Tok / 8, 256, 0, stream>>>(x, wr, gates, H);

    if (ws_size >= needFull) {
        // fast path: pre-convert x + pre-transpose/convert weights to bf16,
        // then fully-async WMMA GEMMs.
        uint16_t* xb  = (uint16_t*)((char*)d_ws + offXb);
        uint16_t* w1t = (uint16_t*)((char*)d_ws + offW1t);  // [E][M][H]
        uint16_t* w2t = (uint16_t*)((char*)d_ws + offW2t);  // [E][H][M]

        cvt_bf16_kernel<<<(Tok * H) / 1024, 256, 0, stream>>>(x, xb);
        transpose_bf16_kernel<<<dim3(M / 32, H / 32, E), 256, 0, stream>>>(w1, w1t, H, M);
        transpose_bf16_kernel<<<dim3(H / 32, M / 32, E), 256, 0, stream>>>(w2, w2t, M, H);

        for (int e = 0; e < E; ++e) {
            moe_gemm<0, true><<<dim3(Tok / 256, M / 128), 256, 0, stream>>>(
                (const void*)xb, (const void*)(w1t + (size_t)e * M * H),
                b1 + (size_t)e * M, nullptr, hid, nullptr, H, M, e);
            moe_gemm<1, true><<<dim3(Tok / 256, H / 128), 256, 0, stream>>>(
                (const void*)hid, (const void*)(w2t + (size_t)e * H * M),
                b2 + (size_t)e * H, gates, nullptr, out, M, H, e);
        }
    } else {
        // fallback: convert inside the GEMMs (needs only gates + hid in ws)
        for (int e = 0; e < E; ++e) {
            moe_gemm<0, false><<<dim3(Tok / 256, M / 128), 256, 0, stream>>>(
                (const void*)x, (const void*)(w1 + (size_t)e * H * M),
                b1 + (size_t)e * M, nullptr, hid, nullptr, H, M, e);
            moe_gemm<1, false><<<dim3(Tok / 256, H / 128), 256, 0, stream>>>(
                (const void*)hid, (const void*)(w2 + (size_t)e * M * H),
                b2 + (size_t)e * H, gates, nullptr, out, M, H, e);
        }
    }
}
